// MultiHeadAttention_76192719832039
// MI455X (gfx1250) — matrix operand order repaired and checked by reading, not compiled, not run on hardware
//
#include <hip/hip_runtime.h>

// ---------------------------------------------------------------------------
// CDNA5 (gfx1250) multi-head causal attention.
// bf16 WMMA (v_wmma_f32_16x16x32_bf16) + flash softmax (exp2 domain, rowsum
// via WMMA) + TDM double-buffered K/V staging (tensor_load_to_lds, TENSORcnt).
// B=4, S=2048, D=512, H=8, dk=64.
// ---------------------------------------------------------------------------

typedef __attribute__((ext_vector_type(16))) __bf16 v16bf;
typedef __attribute__((ext_vector_type(8)))  __bf16 v8bf;
typedef __attribute__((ext_vector_type(8)))  float  v8f;
typedef __attribute__((ext_vector_type(4)))  int    i32x4;
typedef __attribute__((ext_vector_type(8)))  int    i32x8;

#define WMMA_BF16(A, B, C) \
  __builtin_amdgcn_wmma_f32_16x16x32_bf16(false, (A), false, (B), (short)0, (C), false, false)

// Bare v_exp_f32 (args are always <= 0 here; OCML's denorm fix-up is wasted work).
#define EXP2(x) __builtin_amdgcn_exp2f(x)

static __device__ inline v8f zero8() {
  v8f z;
#pragma unroll
  for (int i = 0; i < 8; ++i) z[i] = 0.0f;
  return z;
}

static __device__ inline v16bf pack16(v8bf lo, v8bf hi) {
  v16bf r;
#pragma unroll
  for (int i = 0; i < 8; ++i) { r[i] = lo[i]; r[i + 8] = hi[i]; }
  return r;
}

// A fragment (16x32, M x K): lane m=lane&15, half=lane>>4.
// Element i maps to K = kbase + 8*half + (i<8 ? i : i+8)  -> two contiguous 8-runs.
static __device__ inline v16bf afrag_bf16(const __bf16* row, int kbase, int half) {
  v8bf lo = *(const v8bf*)(row + kbase + 8 * half);
  v8bf hi = *(const v8bf*)(row + kbase + 16 + 8 * half);
  return pack16(lo, hi);
}
// B fragment (32x16, K x N): lane n=lane&15, element i maps to K = kbase + 8*half + (i<8 ? i : i+8).
// `ncol` points at the B^T-stored row for column n (K contiguous).
static __device__ inline v16bf bfrag_bf16(const __bf16* ncol, int kbase, int half) {
  v8bf lo = *(const v8bf*)(ncol + kbase + 8 * half);
  v8bf hi = *(const v8bf*)(ncol + kbase + 16 + 8 * half);
  return pack16(lo, hi);
}

// ---------------------------------------------------------------------------
// Tensor Data Mover: 2-D bf16 tile load Global->LDS (D# per ISA 08 §8.3/8.4).
// Issued from one wave; tracked by TENSORcnt. dims/strides in 2-byte elements.
// ---------------------------------------------------------------------------
static __device__ inline void tdm_load_2d_bf16(uint32_t lds_off, const void* gptr,
                                               uint32_t tensor_d0, uint32_t tensor_d1,
                                               uint32_t tile_d0, uint32_t tile_d1,
                                               uint64_t stride0) {
  const uint64_t ga = (uint64_t)(uintptr_t)gptr;
  i32x4 g0;
  g0[0] = 1;                                   // count=1, user descriptor
  g0[1] = (int)lds_off;                        // LDS byte address (WG-relative)
  g0[2] = (int)(uint32_t)ga;                   // global_addr[31:0]
  g0[3] = (int)(((uint32_t)(ga >> 32) & 0x01FFFFFFu) | 0x80000000u); // [56:32] | type=2
  i32x8 g1;
  g1[0] = 0x00010000;                          // data_size=1 (2 bytes), no flags
  g1[1] = (int)((tensor_d0 & 0xFFFFu) << 16);  // tensor_dim0[15:0]
  g1[2] = (int)(((tensor_d0 >> 16) & 0xFFFFu) | ((tensor_d1 & 0xFFFFu) << 16));
  g1[3] = (int)(((tensor_d1 >> 16) & 0xFFFFu) | ((tile_d0 & 0xFFFFu) << 16));
  g1[4] = (int)(tile_d1 & 0xFFFFu);            // tile_dim1; tile_dim2=0 (unused)
  g1[5] = (int)(uint32_t)(stride0 & 0xFFFFFFFFu);
  g1[6] = (int)(uint32_t)((stride0 >> 32) & 0xFFFFu);  // stride1 = 0
  g1[7] = 0;
  i32x4 gz;
  gz[0] = 0; gz[1] = 0; gz[2] = 0; gz[3] = 0;
  asm volatile("tensor_load_to_lds %0, %1, %2, %3"
               :: "s"(g0), "s"(g1), "s"(gz), "s"(gz)
               : "memory");
}

// ---------------------------------------------------------------------------
// fp32 -> bf16 bulk conversion (one pass; removes cvt work from GEMM loops).
// ---------------------------------------------------------------------------
__global__ __launch_bounds__(256) void cvt_bf16_kernel(const float* __restrict__ in,
                                                       __bf16* __restrict__ out, int n8) {
  const int i = blockIdx.x * 256 + threadIdx.x;
  if (i < n8) {
    v8f x = *(const v8f*)(in + (size_t)i * 8);
    v8bf y;
#pragma unroll
    for (int j = 0; j < 8; ++j) y[j] = (__bf16)x[j];
    *(v8bf*)(out + (size_t)i * 8) = y;
  }
}

// ---------------------------------------------------------------------------
// GEMM: Y = A(8192x512) * W^T, both bf16, W stored [512 x 512] row-major.
// 4 waves / block; each wave computes 32x64 (2 M-tiles x 4 N-tiles):
// 8 WMMAs per 6 fragment loads per 32-wide k-step. Epilogue scaled by oscale.
// MODE 0: fp32 [M,512] row-major | MODE 1: bf16 [B,8,2048,64] | MODE 2: bf16 [B,8,64,2048]
// ---------------------------------------------------------------------------
template <int MODE>
__global__ __launch_bounds__(128) void gemm_bt_kernel(
    const __bf16* __restrict__ A, const __bf16* __restrict__ W, void* __restrict__ Out,
    float oscale) {
  constexpr int Kd = 512, N = 512, S = 2048;
  const int lane = threadIdx.x & 31;
  const int wave = threadIdx.x >> 5;
  const int half = lane >> 4;
  const int lid  = lane & 15;
  const int mbase = blockIdx.x * 128 + wave * 32;
  const int nbase = blockIdx.y * 64;

  const __bf16* Arow0 = A + (size_t)(mbase + lid) * Kd;
  const __bf16* Arow1 = Arow0 + (size_t)16 * Kd;

  v8f acc[2][4];
#pragma unroll
  for (int mt = 0; mt < 2; ++mt)
#pragma unroll
    for (int nt = 0; nt < 4; ++nt) acc[mt][nt] = zero8();

  for (int kk = 0; kk < Kd; kk += 32) {
    const v16bf af0 = afrag_bf16(Arow0, kk, half);
    const v16bf af1 = afrag_bf16(Arow1, kk, half);
#pragma unroll
    for (int nt = 0; nt < 4; ++nt) {
      const __bf16* Wrow = W + (size_t)(nbase + nt * 16 + lid) * Kd;
      const v16bf bf = bfrag_bf16(Wrow, kk, half);
      acc[0][nt] = WMMA_BF16(af0, bf, acc[0][nt]);
      acc[1][nt] = WMMA_BF16(af1, bf, acc[1][nt]);
    }
  }

#pragma unroll
  for (int mt = 0; mt < 2; ++mt) {
#pragma unroll
    for (int nt = 0; nt < 4; ++nt) {
      const int n = nbase + nt * 16 + lid;
#pragma unroll
      for (int r = 0; r < 8; ++r) {
        const int row = mbase + mt * 16 + r + 8 * half;
        const float v = acc[mt][nt][r] * oscale;
        if constexpr (MODE == 0) {
          ((float*)Out)[(size_t)row * N + n] = v;
        } else {
          const int b = row >> 11, s = row & (S - 1);  // S = 2048
          const int h = n >> 6, f = n & 63;
          __bf16* o = (__bf16*)Out;
          if constexpr (MODE == 1)
            o[(((size_t)b * 8 + h) * S + s) * 64 + f] = (__bf16)v;
          else
            o[(((size_t)b * 8 + h) * 64 + f) * (size_t)S + s] = (__bf16)v;
        }
      }
    }
  }
}

// ---------------------------------------------------------------------------
// Flash attention, causal, exp2 domain (scale folded into Q projection).
// Block = 4 waves = 64 queries for one (b,h). 64-key tiles staged via TDM,
// double-buffered. With k0 % 64 == 0 and q0 % 16 == 0, every wave processes
// every tile, and ONLY the final (diagonal) tile needs the causal mask.
// ---------------------------------------------------------------------------
__global__ __launch_bounds__(128) void flash_attn_kernel(
    const __bf16* __restrict__ Q, const __bf16* __restrict__ K,
    const __bf16* __restrict__ Vt, __bf16* __restrict__ Ctx) {
  constexpr int DK = 64, H = 8, S = 2048;
  __shared__ __attribute__((aligned(16))) __bf16 Ks[2][64 * 64];   // [key][feat]
  __shared__ __attribute__((aligned(16))) __bf16 Vs[2][64 * 64];   // [feat][key]
  __shared__ __attribute__((aligned(16))) __bf16 Ps[4][16 * 64];   // per-wave P

  const int tid  = threadIdx.x;
  const int lane = tid & 31;
  const int wave = tid >> 5;
  const int half = lane >> 4;
  const int lid  = lane & 15;
  const int bh   = blockIdx.x;          // b*H + h
  const int qblk = blockIdx.y;          // 64-query block
  const int q0   = qblk * 64 + wave * 16;

  // Scalar (SGPR) wave-0 predicate: forces s_cmp/s_cbranch so only wave 0 ever
  // reaches the TDM issue / TENSORcnt wait (those execute regardless of EXEC).
  const bool is_w0 = ((__builtin_amdgcn_readfirstlane((int)threadIdx.x) >> 5) == 0);

  const __bf16* Qb = Q  + (size_t)bh * S * DK;
  const __bf16* Kb = K  + (size_t)bh * S * DK;
  const __bf16* Vb = Vt + (size_t)bh * DK * S;

  // Q fragments (pre-scaled by 0.125*log2(e) in the projection epilogue).
  const __bf16* Qrow = Qb + (size_t)(q0 + lid) * DK;
  const v16bf qf0 = afrag_bf16(Qrow, 0, half);
  const v16bf qf1 = afrag_bf16(Qrow, 32, half);

  // All-ones B fragment: rowsum(P) = P x ones via WMMA (replaces 32 bpermutes).
  v16bf onesf;
#pragma unroll
  for (int i = 0; i < 16; ++i) onesf[i] = (__bf16)1.0f;

  v8f O[4];
#pragma unroll
  for (int nt = 0; nt < 4; ++nt) O[nt] = zero8();
  float mrow[8], lrow[8];
#pragma unroll
  for (int r = 0; r < 8; ++r) { mrow[r] = -1.0e30f; lrow[r] = 0.0f; }

  auto issue_tile = [&](int kt, int buf) {
    const int k0 = kt * 64;
    // K tile: 64 keys x 64 feat, fully contiguous -> 1-D 4096-element copy.
    tdm_load_2d_bf16((uint32_t)(uintptr_t)&Ks[buf][0], Kb + (size_t)k0 * DK,
                     4096, 1, 4096, 1, 0);
    // V tile: 64 feature rows x 64 keys, row stride S elements.
    tdm_load_2d_bf16((uint32_t)(uintptr_t)&Vs[buf][0], Vb + k0,
                     64, 64, 64, 64, (uint64_t)S);
  };

  const int ktiles = qblk + 1;  // keys 0 .. qblk*64+63; all tiles <= diagonal
  if (is_w0) issue_tile(0, 0);

  for (int kt = 0; kt < ktiles; ++kt) {
    const int buf = kt & 1;
    if (is_w0) __builtin_amdgcn_s_wait_tensorcnt(0);  // tile kt landed
    __syncthreads();
    if (is_w0 && kt + 1 < ktiles) issue_tile(kt + 1, buf ^ 1);  // overlap DMA

    // ---- scores: S = Q * K^T, four 16-key N-tiles, dk = 2 k-steps ----
    v8f s[4];
#pragma unroll
    for (int j = 0; j < 4; ++j) {
      const __bf16* kr = &Ks[buf][(j * 16 + lid) * 64];
      v8f c = zero8();
      c = WMMA_BF16(qf0, bfrag_bf16(kr, 0, half), c);
      c = WMMA_BF16(qf1, bfrag_bf16(kr, 32, half), c);
      s[j] = c;
    }
    // ---- causal mask: only the diagonal tile (uniform branch) ----
    if (kt == ktiles - 1) {
      const int k0 = kt * 64;
#pragma unroll
      for (int j = 0; j < 4; ++j) {
        const int key = k0 + j * 16 + lid;
#pragma unroll
        for (int r = 0; r < 8; ++r) {
          const int q = q0 + r + 8 * half;
          if (key > q) s[j][r] = -1.0e30f;
        }
      }
    }
    // ---- online softmax (exp2 domain); max over 16 owning lanes ----
    float alpha[8];
#pragma unroll
    for (int r = 0; r < 8; ++r) {
      float v = fmaxf(fmaxf(s[0][r], s[1][r]), fmaxf(s[2][r], s[3][r]));
      v = fmaxf(v, __shfl_xor(v, 1, 32));
      v = fmaxf(v, __shfl_xor(v, 2, 32));
      v = fmaxf(v, __shfl_xor(v, 4, 32));
      v = fmaxf(v, __shfl_xor(v, 8, 32));
      const float mn = fmaxf(mrow[r], v);
      alpha[r] = EXP2(mrow[r] - mn);
      mrow[r] = mn;
#pragma unroll
      for (int j = 0; j < 4; ++j) s[j][r] = EXP2(s[j][r] - mn);
    }
    // ---- D-fragment -> A-fragment layout change through per-wave LDS ----
    __bf16* P = &Ps[wave][0];
#pragma unroll
    for (int j = 0; j < 4; ++j)
#pragma unroll
      for (int r = 0; r < 8; ++r)
        P[(r + 8 * half) * 64 + j * 16 + lid] = (__bf16)s[j][r];
    const v16bf pf0 = afrag_bf16(P + lid * 64, 0, half);
    const v16bf pf1 = afrag_bf16(P + lid * 64, 32, half);
    // ---- rowsum via WMMA: l = l*alpha + P x ones ----
    v8f sm = zero8();
    sm = WMMA_BF16(pf0, onesf, sm);
    sm = WMMA_BF16(pf1, onesf, sm);
#pragma unroll
    for (int r = 0; r < 8; ++r) lrow[r] = lrow[r] * alpha[r] + sm[r];
#pragma unroll
    for (int nt = 0; nt < 4; ++nt)
#pragma unroll
      for (int r = 0; r < 8; ++r) O[nt][r] *= alpha[r];
    // ---- O += P * V  (4 feature N-tiles x 2 k-steps) ----
#pragma unroll
    for (int nt = 0; nt < 4; ++nt) {
      const __bf16* vr = &Vs[buf][(nt * 16 + lid) * 64];
      O[nt] = WMMA_BF16(pf0, bfrag_bf16(vr, 0, half), O[nt]);
      O[nt] = WMMA_BF16(pf1, bfrag_bf16(vr, 32, half), O[nt]);
    }
    __syncthreads();  // everyone done with buf before it is refilled at kt+2
  }

  // ---- epilogue: normalize, write context bf16 [B,S,H*64] ----
  float inv[8];
#pragma unroll
  for (int r = 0; r < 8; ++r) inv[r] = 1.0f / lrow[r];
  const int b = bh / H, h = bh % H;
#pragma unroll
  for (int nt = 0; nt < 4; ++nt) {
#pragma unroll
    for (int r = 0; r < 8; ++r) {
      const int q = q0 + r + 8 * half;
      Ctx[((size_t)b * S + q) * 512 + h * 64 + nt * 16 + lid] =
          (__bf16)(O[nt][r] * inv[r]);
    }
  }
}

// ---------------------------------------------------------------------------
extern "C" void kernel_launch(void* const* d_in, const int* in_sizes, int n_in,
                              void* d_out, int out_size, void* d_ws, size_t ws_size,
                              hipStream_t stream) {
  (void)in_sizes; (void)n_in; (void)out_size; (void)ws_size;
  const float* inQ = (const float*)d_in[0];
  const float* inK = (const float*)d_in[1];
  const float* inV = (const float*)d_in[2];
  const float* WQ  = (const float*)d_in[3];
  const float* WK  = (const float*)d_in[4];
  const float* WV  = (const float*)d_in[5];
  const float* WO  = (const float*)d_in[6];

  const int B = 4, S = 2048, D = 512, H = 8;
  const int M = B * S;  // 8192

  const size_t nX = (size_t)M * D;   // 4,194,304 elements per activation tensor
  const size_t nW = (size_t)D * D;   // 262,144 elements per weight

  __bf16* p   = (__bf16*)d_ws;
  __bf16* Xq  = p; p += nX;
  __bf16* Xk  = p; p += nX;
  __bf16* Xv  = p; p += nX;
  __bf16* Wqb = p; p += nW;
  __bf16* Wkb = p; p += nW;
  __bf16* Wvb = p; p += nW;
  __bf16* Wob = p; p += nW;
  __bf16* Qbf = p; p += nX;
  __bf16* Kbf = p; p += nX;
  __bf16* Vtb = p; p += nX;
  __bf16* Ctx = p; p += nX;

  // One-shot fp32 -> bf16 conversions (memory-bound; trivial at 23.3 TB/s).
  cvt_bf16_kernel<<<(int)(nX / 8 / 256), 256, 0, stream>>>(inQ, Xq, (int)(nX / 8));
  cvt_bf16_kernel<<<(int)(nX / 8 / 256), 256, 0, stream>>>(inK, Xk, (int)(nX / 8));
  cvt_bf16_kernel<<<(int)(nX / 8 / 256), 256, 0, stream>>>(inV, Xv, (int)(nX / 8));
  cvt_bf16_kernel<<<(int)(nW / 8 / 256), 256, 0, stream>>>(WQ, Wqb, (int)(nW / 8));
  cvt_bf16_kernel<<<(int)(nW / 8 / 256), 256, 0, stream>>>(WK, Wkb, (int)(nW / 8));
  cvt_bf16_kernel<<<(int)(nW / 8 / 256), 256, 0, stream>>>(WV, Wvb, (int)(nW / 8));
  cvt_bf16_kernel<<<(int)(nW / 8 / 256), 256, 0, stream>>>(WO, Wob, (int)(nW / 8));

  dim3 gg(M / 128, D / 64);  // 64 x 8 blocks, 128 threads (4 waves, 32x64 each)
  // Q projection folds the softmax scale 1/sqrt(64) and the exp2 conversion
  // factor log2(e); scores then feed exp2 directly.
  const float qscale = 0.125f * 1.4426950408889634f;
  gemm_bt_kernel<1><<<gg, 128, 0, stream>>>(Xq, Wqb, Qbf, qscale);
  gemm_bt_kernel<1><<<gg, 128, 0, stream>>>(Xk, Wkb, Kbf, 1.0f);
  gemm_bt_kernel<2><<<gg, 128, 0, stream>>>(Xv, Wvb, Vtb, 1.0f);

  flash_attn_kernel<<<dim3(B * H, S / 64), 128, 0, stream>>>(Qbf, Kbf, Vtb, Ctx);

  gemm_bt_kernel<0><<<gg, 128, 0, stream>>>(Ctx, Wob, d_out, 1.0f);
}
